// CrystalDynamics_77979426226215
// MI455X (gfx1250) — matrix operand order repaired and checked by reading, not compiled, not run on hardware
//
#include <hip/hip_runtime.h>
#include <cstdint>
#include <cstddef>

typedef _Float16 v16h __attribute__((ext_vector_type(16)));
typedef _Float16 v8h  __attribute__((ext_vector_type(8)));
typedef float    v8f  __attribute__((ext_vector_type(8)));

namespace cd {

constexpr int N  = 20000;   // nodes
constexpr int G  = 400;     // graphs
constexpr int E  = 240000;  // edges (12 per node, contiguous by dst)
constexpr int ND = 128;
constexpr int NB = N / 16;  // 1250 blocks of 16 nodes

__device__ __forceinline__ float silu_f(float x) { return x / (1.0f + __expf(-x)); }
__device__ __forceinline__ v8h ld8(const _Float16* p) { return *(const v8h*)p; }
__device__ __forceinline__ v16h mk16(v8h a, v8h b) {
  return __builtin_shufflevector(a, b, 0,1,2,3,4,5,6,7,8,9,10,11,12,13,14,15);
}
// A operand 16x32 f16: lane<16 holds K{0..7,16..23}, lane>=16 holds K{8..15,24..31}
__device__ __forceinline__ v16h load_a(const _Float16* rowp_k0, int hi) {
  return mk16(ld8(rowp_k0 + hi * 8), ld8(rowp_k0 + 16 + hi * 8));
}
// B operand 32x16 f16 from transposed weights Wt[outCol][K]: lane<16 holds K{0..7,16..23}, lane>=16 holds K{8..15,24..31}
__device__ __forceinline__ v16h load_b(const _Float16* rowp, int k0, int hi) {
  const _Float16* p = rowp + k0 + hi * 8;
  return mk16(ld8(p), ld8(p + 16));
}
__device__ __forceinline__ v8f wmma(v16h a, v16h b, v8f c) {
  return __builtin_amdgcn_wmma_f32_16x16x32_f16(false, a, false, b, (short)0, c, false, false);
}

// ---------------- small prep kernels ----------------

__global__ void k_zero(float* p, int n) {
  int i = blockIdx.x * blockDim.x + threadIdx.x;
  if (i < n) p[i] = 0.f;
}

// time MLP: temb[g] = silu(sinusoid @ W1 + b1) @ W2 + b2   (G=400, tiny -> scalar fp32)
__global__ __launch_bounds__(256)
void k_tmlp(const float* __restrict__ t, const float* __restrict__ W1, const float* __restrict__ b1,
            const float* __restrict__ W2, const float* __restrict__ b2, float* __restrict__ temb) {
  __shared__ float s_in[128];
  __shared__ float s_hid[256];
  const int g = blockIdx.x, j = threadIdx.x;
  if (j < 64) {
    float fr = __expf(-0.14619588920256570f * (float)j);  // -ln(10000)/63
    float ang = t[g] * fr;
    s_in[j] = sinf(ang);
    s_in[j + 64] = cosf(ang);
  }
  __syncthreads();
  float a = b1[j];
  for (int k = 0; k < 128; ++k) a += s_in[k] * W1[k * 256 + j];
  s_hid[j] = silu_f(a);
  __syncthreads();
  if (j < 128) {
    float a2 = b2[j];
    for (int k = 0; k < 256; ++k) a2 += s_hid[k] * W2[k * 128 + j];
    temb[g * 128 + j] = a2;
  }
}

// t16[n][c] = f16(temb[batch[n]][c])
__global__ void k_t16(const float* __restrict__ temb, const int* __restrict__ batch,
                      _Float16* __restrict__ t16) {
  int i = blockIdx.x * blockDim.x + threadIdx.x;   // exact grid N*128
  int n = i >> 7, c = i & 127;
  t16[i] = (_Float16)temb[batch[n] * 128 + c];
}

// x0[n][0:160] = f16([z[n] ; species_emb[species[n]]])
__global__ void k_x0(const float* __restrict__ z, const int* __restrict__ species,
                     const float* __restrict__ spemb, _Float16* __restrict__ x0) {
  int i = blockIdx.x * blockDim.x + threadIdx.x;   // exact grid N*160
  int n = i / 160, k = i - n * 160;
  float v = (k < 128) ? z[n * 128 + k] : spemb[species[n] * 32 + (k - 128)];
  x0[i] = (_Float16)v;
}

// per-edge geometry: dist_sq and diff/dist
__global__ void k_geom(const float* __restrict__ coords, const int* __restrict__ esrc,
                       const int* __restrict__ edst, float* __restrict__ dsq,
                       float* __restrict__ inv) {
  int e = blockIdx.x * blockDim.x + threadIdx.x;
  if (e >= E) return;
  int s = esrc[e], d = edst[e];
  float dx = coords[s * 3 + 0] - coords[d * 3 + 0];
  float dy = coords[s * 3 + 1] - coords[d * 3 + 1];
  float dz = coords[s * 3 + 2] - coords[d * 3 + 2];
  float q = dx * dx + dy * dy + dz * dz;
  float ir = 1.f / sqrtf(q + 1e-8f);
  dsq[e] = q;
  inv[e * 3 + 0] = dx * ir;
  inv[e * 3 + 1] = dy * ir;
  inv[e * 3 + 2] = dz * ir;
}

// transpose+convert weights: dst[n][k] = f16(src[k<skip ? k : k+1][n])  (src row-major [K_src][Nout])
__global__ void k_wt(const float* __restrict__ src, _Float16* __restrict__ dst,
                     int K, int Nout, int skip) {
  int i = blockIdx.x * blockDim.x + threadIdx.x;
  if (i >= K * Nout) return;
  int k = i / Nout, n = i - k * Nout;
  int sk = (k < skip) ? k : k + 1;
  dst[(size_t)n * K + k] = (_Float16)src[(size_t)sk * Nout + n];
}

// ---------------- node embedding: h0 = x0 @ ne_W + b (K=160) ----------------
__global__ __launch_bounds__(128)
void k_embed(const _Float16* __restrict__ x0, const _Float16* __restrict__ Wt,
             const float* __restrict__ bias, float* __restrict__ h32, _Float16* __restrict__ h16) {
  const int lane = threadIdx.x & 31, wave = threadIdx.x >> 5;
  const int hi = lane >> 4, r = lane & 15;
  const int node0 = blockIdx.x * 16;
  const _Float16* arow = x0 + (size_t)(node0 + r) * 160;
  v16h A[5];
#pragma unroll
  for (int kk = 0; kk < 5; ++kk) A[kk] = load_a(arow + kk * 32, hi);
  const v8f zero8 = {0.f,0.f,0.f,0.f,0.f,0.f,0.f,0.f};
#pragma unroll
  for (int w = 0; w < 2; ++w) {
    const int nt = wave * 2 + w;
    const _Float16* brow = Wt + (size_t)(nt * 16 + r) * 160;
    v8f acc = zero8;
#pragma unroll
    for (int kk = 0; kk < 5; ++kk) acc = wmma(A[kk], load_b(brow, kk * 32, hi), acc);
    const int ncol = nt * 16 + r;
    const float b = bias[ncol];
#pragma unroll
    for (int q = 0; q < 8; ++q) {
      const int row = q + 8 * hi;
      float v = acc[q] + b;
      h32[(size_t)(node0 + row) * ND + ncol] = v;
      h16[(size_t)(node0 + row) * ND + ncol] = (_Float16)v;
    }
  }
}

// ---------------- fused layer: edge MLPs + gate + segment sums + node MLP ----------------
// Block = 16 nodes + their 192 incoming edges (dst-contiguous, K=12). 128 threads = 4 waves.
__global__ __launch_bounds__(128)
void k_layer(const float* __restrict__ h32_in, const _Float16* __restrict__ h16_in,
             float* __restrict__ h32_out, _Float16* __restrict__ h16_out,
             const _Float16* __restrict__ t16,
             const float* __restrict__ dsq, const float* __restrict__ inv,
             const int* __restrict__ esrc, const int* __restrict__ edst,
             const _Float16* __restrict__ eW1t, const float* __restrict__ wdist,
             const float* __restrict__ eb1,
             const _Float16* __restrict__ eW2t, const float* __restrict__ eb2,
             const _Float16* __restrict__ cW1t, const float* __restrict__ cb1,
             const float* __restrict__ cw2,
             const _Float16* __restrict__ nW1t, const float* __restrict__ nb1,
             const _Float16* __restrict__ nW2t, const float* __restrict__ nb2,
             float* __restrict__ shift_out) {
  __shared__ float    s_dsq[192];
  __shared__ float    s_inv[192 * 3];
  __shared__ int      s_dl[192];
  __shared__ _Float16 s_stage[4][16 * ND];   // per-wave m1/m2 re-staging (C-layout -> A-layout)
  __shared__ float    s_mi[16 * ND];
  __shared__ _Float16 s_mi16[16 * ND];
  __shared__ float    s_shift[48];
  __shared__ _Float16 s_u[16 * ND];

  const int tid = threadIdx.x;
  const int lane = tid & 31, wave = tid >> 5;
  const int hi = lane >> 4, r = lane & 15;
  const int node0 = blockIdx.x * 16;
  const int e0 = node0 * 12;
  const v8f zero8 = {0.f,0.f,0.f,0.f,0.f,0.f,0.f,0.f};

  for (int i = tid; i < 192; i += 128) {
    s_dsq[i] = dsq[e0 + i];
    s_dl[i] = edst[e0 + i] - node0;
    s_inv[i * 3 + 0] = inv[(e0 + i) * 3 + 0];
    s_inv[i * 3 + 1] = inv[(e0 + i) * 3 + 1];
    s_inv[i * 3 + 2] = inv[(e0 + i) * 3 + 2];
  }
  for (int i = tid; i < 16 * ND; i += 128) s_mi[i] = 0.f;
  if (tid < 48) s_shift[tid] = 0.f;
  __syncthreads();

  _Float16* st = &s_stage[wave][0];
  for (int tile = wave; tile < 12; tile += 4) {          // uniform per wave
    const int le = tile * 16 + r;
    const int eg = e0 + le;
    const int sn = esrc[eg];
    const int dn = edst[eg];
    // A rows: [h_src(0:128) ; h_dst(128:256) ; t_src(256:384)], dist_sq handled as fp32 rank-1
    const _Float16* aseg[3] = { h16_in + (size_t)sn * ND,
                                h16_in + (size_t)dn * ND,
                                t16 + (size_t)sn * ND };
    v8f acc[8];
#pragma unroll
    for (int n = 0; n < 8; ++n) acc[n] = zero8;
#pragma unroll
    for (int kk = 0; kk < 12; ++kk) {
      const int k0 = kk * 32;
      v16h A = load_a(aseg[k0 >> 7] + (k0 & 127), hi);
#pragma unroll
      for (int n = 0; n < 8; ++n)
        acc[n] = wmma(A, load_b(eW1t + (size_t)(n * 16 + r) * 384, k0, hi), acc[n]);
    }
    // + dist_sq * eW1[row 256] + bias, silu, stage m1 as f16
#pragma unroll
    for (int n = 0; n < 8; ++n) {
      const int ncol = n * 16 + r;
      const float wd = wdist[ncol], b = eb1[ncol];
#pragma unroll
      for (int q = 0; q < 8; ++q) {
        const int row = q + 8 * hi;
        float x = acc[n][q] + s_dsq[tile * 16 + row] * wd + b;
        st[row * ND + ncol] = (_Float16)silu_f(x);
      }
    }
    // m2 = silu(m1 @ eW2 + eb2); accumulate m_i; restage m2
    {
      const _Float16* arow = st + r * ND;
      v16h A2[4];
#pragma unroll
      for (int kk = 0; kk < 4; ++kk) A2[kk] = load_a(arow + kk * 32, hi);
#pragma unroll
      for (int n = 0; n < 8; ++n) {
        v8f a2 = zero8;
#pragma unroll
        for (int kk = 0; kk < 4; ++kk)
          a2 = wmma(A2[kk], load_b(eW2t + (size_t)(n * 16 + r) * ND, kk * 32, hi), a2);
        const int ncol = n * 16 + r;
        const float b = eb2[ncol];
#pragma unroll
        for (int q = 0; q < 8; ++q) {
          const int row = q + 8 * hi;
          const int lrow = tile * 16 + row;
          float x = silu_f(a2[q] + b);
          atomicAdd(&s_mi[s_dl[lrow] * ND + ncol], x);
          st[row * ND + ncol] = (_Float16)x;
        }
      }
    }
    // gate = silu(m2 @ cW1 + cb1) @ cW2 ; shift accumulation (LDS only)
    {
      const _Float16* arow = st + r * ND;
      v16h A3[4];
#pragma unroll
      for (int kk = 0; kk < 4; ++kk) A3[kk] = load_a(arow + kk * 32, hi);
      float pg[8] = {0.f,0.f,0.f,0.f,0.f,0.f,0.f,0.f};
#pragma unroll
      for (int n = 0; n < 8; ++n) {
        v8f a3 = zero8;
#pragma unroll
        for (int kk = 0; kk < 4; ++kk)
          a3 = wmma(A3[kk], load_b(cW1t + (size_t)(n * 16 + r) * ND, kk * 32, hi), a3);
        const int ncol = n * 16 + r;
        const float b = cb1[ncol], w2 = cw2[ncol];
#pragma unroll
        for (int q = 0; q < 8; ++q) pg[q] += silu_f(a3[q] + b) * w2;
      }
      // reduce over the 16 lanes of each half (xor masks stay within halves)
#pragma unroll
      for (int m = 1; m < 16; m <<= 1) {
#pragma unroll
        for (int q = 0; q < 8; ++q) pg[q] += __shfl_xor(pg[q], m, 32);
      }
      if (r == 0) {
#pragma unroll
        for (int q = 0; q < 8; ++q) {
          const int lrow = tile * 16 + q + 8 * hi;
          const float gv = pg[q];
          const int nl = s_dl[lrow];
          atomicAdd(&s_shift[nl * 3 + 0], s_inv[lrow * 3 + 0] * gv);
          atomicAdd(&s_shift[nl * 3 + 1], s_inv[lrow * 3 + 1] * gv);
          atomicAdd(&s_shift[nl * 3 + 2], s_inv[lrow * 3 + 2] * gv);
        }
      }
    }
  }
  __syncthreads();
  for (int i = tid; i < 16 * ND; i += 128) s_mi16[i] = (_Float16)s_mi[i];
  if (tid < 48) shift_out[node0 * 3 + tid] += s_shift[tid];   // block owns nodes exclusively
  __syncthreads();

  // node update: h += silu([h ; m_i ; t] @ nW1 + nb1) @ nW2 + nb2
  {
    const _Float16* aseg[3] = { h16_in + (size_t)(node0 + r) * ND,
                                (const _Float16*)s_mi16 + (size_t)r * ND,
                                t16 + (size_t)(node0 + r) * ND };
    v16h A[12];
#pragma unroll
    for (int kk = 0; kk < 12; ++kk) {
      const int k0 = kk * 32;
      A[kk] = load_a(aseg[k0 >> 7] + (k0 & 127), hi);
    }
#pragma unroll
    for (int w = 0; w < 2; ++w) {
      const int nt = wave * 2 + w;
      v8f acc = zero8;
#pragma unroll
      for (int kk = 0; kk < 12; ++kk)
        acc = wmma(A[kk], load_b(nW1t + (size_t)(nt * 16 + r) * 384, kk * 32, hi), acc);
      const int ncol = nt * 16 + r;
      const float b = nb1[ncol];
#pragma unroll
      for (int q = 0; q < 8; ++q)
        s_u[(q + 8 * hi) * ND + ncol] = (_Float16)silu_f(acc[q] + b);
    }
    __syncthreads();
    const _Float16* arow = (const _Float16*)s_u + (size_t)r * ND;
    v16h A2[4];
#pragma unroll
    for (int kk = 0; kk < 4; ++kk) A2[kk] = load_a(arow + kk * 32, hi);
#pragma unroll
    for (int w = 0; w < 2; ++w) {
      const int nt = wave * 2 + w;
      v8f acc = zero8;
#pragma unroll
      for (int kk = 0; kk < 4; ++kk)
        acc = wmma(A2[kk], load_b(nW2t + (size_t)(nt * 16 + r) * ND, kk * 32, hi), acc);
      const int ncol = nt * 16 + r;
      const float b = nb2[ncol];
#pragma unroll
      for (int q = 0; q < 8; ++q) {
        const int node = node0 + q + 8 * hi;
        float v = h32_in[(size_t)node * ND + ncol] + acc[q] + b;
        h32_out[(size_t)node * ND + ncol] = v;
        h16_out[(size_t)node * ND + ncol] = (_Float16)v;
      }
    }
  }
}

} // namespace cd

extern "C" void kernel_launch(void* const* d_in, const int* in_sizes, int n_in,
                              void* d_out, int out_size, void* d_ws, size_t ws_size,
                              hipStream_t stream) {
  using namespace cd;
  (void)in_sizes; (void)n_in; (void)out_size; (void)ws_size;

  const float* z       = (const float*)d_in[0];
  const float* t       = (const float*)d_in[1];
  const float* coords  = (const float*)d_in[2];
  const int*   species = (const int*)d_in[3];
  const int*   batch   = (const int*)d_in[4];
  const int*   esrc    = (const int*)d_in[5];
  const int*   edst    = (const int*)d_in[6];
  const float* spemb   = (const float*)d_in[7];
  const float* tmW1    = (const float*)d_in[8];
  const float* tmb1    = (const float*)d_in[9];
  const float* tmW2    = (const float*)d_in[10];
  const float* tmb2    = (const float*)d_in[11];
  const float* neW     = (const float*)d_in[12];
  const float* neb     = (const float*)d_in[13];
  const float* eW1     = (const float*)d_in[14];
  const float* eb1     = (const float*)d_in[15];
  const float* eW2     = (const float*)d_in[16];
  const float* eb2     = (const float*)d_in[17];
  const float* cW1     = (const float*)d_in[18];
  const float* cb1     = (const float*)d_in[19];
  const float* cW2     = (const float*)d_in[20];
  const float* nW1     = (const float*)d_in[21];
  const float* nb1     = (const float*)d_in[22];
  const float* nW2     = (const float*)d_in[23];
  const float* nb2     = (const float*)d_in[24];
  float* out = (float*)d_out;

  // deterministic bump allocation in workspace (same layout every call)
  char* w = (char*)d_ws;
  auto alloc = [&](size_t b) { void* p = (void*)w; w += (b + 255) & ~(size_t)255; return p; };
  float*    temb = (float*)alloc((size_t)G * 128 * 4);
  _Float16* t16  = (_Float16*)alloc((size_t)N * 128 * 2);
  _Float16* x0   = (_Float16*)alloc((size_t)N * 160 * 2);
  float*    dsq  = (float*)alloc((size_t)E * 4);
  float*    inv  = (float*)alloc((size_t)E * 12);
  float*    hA32 = (float*)alloc((size_t)N * 128 * 4);
  float*    hB32 = (float*)alloc((size_t)N * 128 * 4);
  _Float16* hA16 = (_Float16*)alloc((size_t)N * 128 * 2);
  _Float16* hB16 = (_Float16*)alloc((size_t)N * 128 * 2);
  _Float16* neWt = (_Float16*)alloc((size_t)160 * 128 * 2);
  _Float16* eW1t = (_Float16*)alloc((size_t)4 * 128 * 384 * 2);
  _Float16* eW2t = (_Float16*)alloc((size_t)4 * 128 * 128 * 2);
  _Float16* cW1t = (_Float16*)alloc((size_t)4 * 128 * 128 * 2);
  _Float16* nW1t = (_Float16*)alloc((size_t)4 * 128 * 384 * 2);
  _Float16* nW2t = (_Float16*)alloc((size_t)4 * 128 * 128 * 2);

  k_zero<<<(3 * N + 255) / 256, 256, 0, stream>>>(out, 3 * N);
  k_tmlp<<<G, 256, 0, stream>>>(t, tmW1, tmb1, tmW2, tmb2, temb);
  k_t16<<<(N * 128) / 256, 256, 0, stream>>>(temb, batch, t16);
  k_x0<<<(N * 160) / 256, 256, 0, stream>>>(z, species, spemb, x0);
  k_geom<<<(E + 255) / 256, 256, 0, stream>>>(coords, esrc, edst, dsq, inv);

  k_wt<<<(160 * 128 + 255) / 256, 256, 0, stream>>>(neW, neWt, 160, 128, 160);
  for (int l = 0; l < 4; ++l) {
    // eW1: rows 0..255 (h_src,h_dst) + rows 257..384 (t_edges); row 256 (dist_sq) handled as rank-1
    k_wt<<<(384 * 128) / 256, 256, 0, stream>>>(eW1 + (size_t)l * 385 * 128,
                                                eW1t + (size_t)l * 128 * 384, 384, 128, 256);
    k_wt<<<(128 * 128) / 256, 256, 0, stream>>>(eW2 + (size_t)l * 128 * 128,
                                                eW2t + (size_t)l * 128 * 128, 128, 128, 128);
    k_wt<<<(128 * 128) / 256, 256, 0, stream>>>(cW1 + (size_t)l * 128 * 128,
                                                cW1t + (size_t)l * 128 * 128, 128, 128, 128);
    k_wt<<<(384 * 128) / 256, 256, 0, stream>>>(nW1 + (size_t)l * 384 * 128,
                                                nW1t + (size_t)l * 128 * 384, 384, 128, 384);
    k_wt<<<(128 * 128) / 256, 256, 0, stream>>>(nW2 + (size_t)l * 128 * 128,
                                                nW2t + (size_t)l * 128 * 128, 128, 128, 128);
  }

  k_embed<<<NB, 128, 0, stream>>>(x0, neWt, neb, hA32, hA16);

  for (int l = 0; l < 4; ++l) {
    const float*    hi32 = (l & 1) ? hB32 : hA32;
    const _Float16* hi16 = (l & 1) ? hB16 : hA16;
    float*    ho32 = (l == 3) ? (out + (size_t)3 * N) : ((l & 1) ? hA32 : hB32);
    _Float16* ho16 = (l & 1) ? hA16 : hB16;
    k_layer<<<NB, 128, 0, stream>>>(
        hi32, hi16, ho32, ho16, t16, dsq, inv, esrc, edst,
        eW1t + (size_t)l * 128 * 384, eW1 + ((size_t)l * 385 + 256) * 128, eb1 + l * 128,
        eW2t + (size_t)l * 128 * 128, eb2 + l * 128,
        cW1t + (size_t)l * 128 * 128, cb1 + l * 128, cW2 + l * 128,
        nW1t + (size_t)l * 128 * 384, nb1 + l * 128,
        nW2t + (size_t)l * 128 * 128, nb2 + l * 128,
        out);
  }
}
